// VQVAE_25503515804101
// MI455X (gfx1250) — matrix operand order repaired and checked by reading, not compiled, not run on hardware
//
#include <hip/hip_runtime.h>

// ---------------- types ----------------
typedef __attribute__((ext_vector_type(16))) __bf16 v16bf;
typedef __attribute__((ext_vector_type(8)))  float  v8f;
typedef __attribute__((ext_vector_type(4)))  int    vi4;

struct __attribute__((aligned(16))) U4 { unsigned int x, y, z, w; };
struct __attribute__((aligned(16))) F4 { float x, y, z, w; };

union FragB { U4 q[2]; v16bf v; };   // 32 bytes = 16 bf16 = one WMMA A/B operand

__device__ __forceinline__ unsigned short f2bf(float f) {
    union { float f; unsigned int u; } c; c.f = f;
    unsigned int u = c.u;
    unsigned int r = u + 0x7FFFu + ((u >> 16) & 1u);
    return (unsigned short)(r >> 16);
}

// ---------------- CDNA5 async data movement helpers ----------------
// Async global->LDS copy (ASYNCcnt), 16 bytes per lane.
__device__ __forceinline__ void async_cp16(unsigned short* ldsBase, unsigned short* l,
                                           const unsigned short* g) {
#if __has_builtin(__builtin_amdgcn_global_load_async_to_lds_b128)
    __builtin_amdgcn_global_load_async_to_lds_b128(
        (__attribute__((address_space(1))) vi4*)(unsigned short*)g,
        (__attribute__((address_space(3))) vi4*)l, 0, 0);
#else
    unsigned int off = (unsigned int)((const char*)l - (const char*)ldsBase);
    asm volatile("global_load_async_to_lds_b128 %0, %1, off"
                 :: "v"(off), "v"((unsigned long long)g) : "memory");
#endif
}

__device__ __forceinline__ void wait_async0() {
#if __has_builtin(__builtin_amdgcn_s_wait_asynccnt)
    __builtin_amdgcn_s_wait_asynccnt(0);
#else
    asm volatile("s_wait_asynccnt 0x0" ::: "memory");
#endif
}

__device__ __forceinline__ void wait_tensor0() {
#if __has_builtin(__builtin_amdgcn_s_wait_tensorcnt)
    __builtin_amdgcn_s_wait_tensorcnt(0);
#else
    asm volatile("s_wait_tensorcnt 0x0" ::: "memory");
#endif
}

#if __has_builtin(__builtin_amdgcn_tensor_load_to_lds)
#define HAVE_TDM 1
typedef __attribute__((ext_vector_type(4))) unsigned int u32x4;
typedef __attribute__((ext_vector_type(8))) int i32x8;
typedef __attribute__((ext_vector_type(4))) int i32x4;

// 2D TDM tile load: tileH rows x tileW elems (2B each) from a row-major tensor with
// row stride `strideElems`, into LDS at byte offset `ldsByteOff`, LDS rows padded via
// pad_interval/pad_amount (DWORD-coded per D# spec).
__device__ __forceinline__ void tdm_load_2d(unsigned int ldsByteOff, const void* gaddr,
                                            unsigned int tileW, unsigned int tileH,
                                            unsigned int tensorW, unsigned int tensorH,
                                            unsigned int strideElems,
                                            unsigned int padInterval, unsigned int padAmount) {
    unsigned long long ga = (unsigned long long)gaddr;
    u32x4 g0;
    g0[0] = 1u;                                                      // count=1, user mode
    g0[1] = ldsByteOff;                                              // lds_addr
    g0[2] = (unsigned int)ga;                                        // global_addr[31:0]
    g0[3] = (unsigned int)((ga >> 32) & 0x1FFFFFFull) | (2u << 30);  // global_addr[56:32], type=2
    i32x8 g1;
    g1[0] = (int)((1u << 16) | (1u << 20) |                          // data_size=2B, pad_enable
                  (padInterval << 22) | (padAmount << 25));
    g1[1] = (int)((tensorW & 0xFFFFu) << 16);                        // tensor_dim0[15:0]
    g1[2] = (int)(((tensorW >> 16) & 0xFFFFu) | ((tensorH & 0xFFFFu) << 16));
    g1[3] = (int)(((tensorH >> 16) & 0xFFFFu) | ((tileW & 0xFFFFu) << 16)); // tile_dim0
    g1[4] = (int)(tileH & 0xFFFFu);                                  // tile_dim1 (tile_dim2=0)
    g1[5] = (int)strideElems;                                        // tensor_dim0_stride[31:0]
    g1[6] = 0; g1[7] = 0;
    i32x4 z4 = {0, 0, 0, 0};
    i32x8 z8 = {0, 0, 0, 0, 0, 0, 0, 0};
    __builtin_amdgcn_tensor_load_to_lds(g0, g1, z4, z4, z8, 0);
}
#endif

// ---------------- problem constants ----------------
#define NROWS 32768
#define INDIM 1024
#define EDIM  512
#define KCODES 8192
#define BETA 0.25f

// ---------------- small utility kernels ----------------
__global__ void f2bf_kernel(const float* __restrict__ s,
                            unsigned short* __restrict__ d, long n) {
    long i = ((long)blockIdx.x * blockDim.x + threadIdx.x) * 8;
    if (i >= n) return;
    F4 a = *(const F4*)(s + i);
    F4 b = *(const F4*)(s + i + 4);
    union { unsigned short h[8]; U4 q; } u;
    u.h[0] = f2bf(a.x); u.h[1] = f2bf(a.y); u.h[2] = f2bf(a.z); u.h[3] = f2bf(a.w);
    u.h[4] = f2bf(b.x); u.h[5] = f2bf(b.y); u.h[6] = f2bf(b.z); u.h[7] = f2bf(b.w);
    *(U4*)(d + i) = u.q;
}

__global__ void transpose_f2bf_kernel(const float* __restrict__ s,
                                      unsigned short* __restrict__ d,
                                      int rows, int cols) {
    __shared__ float tile[32][33];
    int ctiles = cols >> 5;
    int bx = blockIdx.x % ctiles;
    int by = blockIdx.x / ctiles;
    int tx = threadIdx.x & 31, ty = threadIdx.x >> 5;
    #pragma unroll
    for (int i = 0; i < 32; i += 8)
        tile[ty + i][tx] = s[(size_t)(by * 32 + ty + i) * cols + bx * 32 + tx];
    __syncthreads();
    #pragma unroll
    for (int i = 0; i < 32; i += 8)
        d[(size_t)(bx * 32 + ty + i) * rows + by * 32 + tx] = f2bf(tile[tx][ty + i]);
}

__global__ void cbnorm_kernel(const float* __restrict__ cb, float* __restrict__ cbn) {
    int row = blockIdx.x * 8 + (threadIdx.x >> 5);
    int lane = threadIdx.x & 31;
    float acc = 0.f;
    #pragma unroll
    for (int i = 0; i < 16; ++i) {
        float v = cb[(size_t)row * EDIM + lane + i * 32];
        acc += v * v;
    }
    #pragma unroll
    for (int m = 16; m >= 1; m >>= 1) acc += __shfl_xor(acc, m, 32);
    if (lane == 0) cbn[row] = acc;
}

__global__ void zero_kernel(float* p) { *p = 0.f; }

__global__ void finalize_kernel(const float* __restrict__ lossAcc, float* __restrict__ out) {
    *out = lossAcc[0] * (1.0f + BETA) / ((float)NROWS * (float)EDIM);
}

__global__ void gather_loss_kernel(const int* __restrict__ idx,
                                   const float* __restrict__ cbF,
                                   const unsigned short* __restrict__ cbBf,
                                   const float* __restrict__ zF,
                                   unsigned short* __restrict__ zqBf,
                                   float* __restrict__ lossAcc) {
    int row = blockIdx.x * 8 + (threadIdx.x >> 5);
    int lane = threadIdx.x & 31;
    int k = idx[row];
    const U4* s = (const U4*)(cbBf + (size_t)k * EDIM);
    U4* d = (U4*)(zqBf + (size_t)row * EDIM);
    d[lane] = s[lane];
    d[lane + 32] = s[lane + 32];
    float acc = 0.f;
    #pragma unroll
    for (int i = 0; i < 16; ++i) {
        int e = lane + i * 32;
        float diff = cbF[(size_t)k * EDIM + e] - zF[(size_t)row * EDIM + e];
        acc += diff * diff;
    }
    #pragma unroll
    for (int m = 16; m >= 1; m >>= 1) acc += __shfl_xor(acc, m, 32);
    if (lane == 0) atomicAdd(lossAcc, acc);
}

// ---------------- main WMMA GEMM: out = act(A[M,K] * BT[N,K]^T + bias) ----------------
// 128x128x32 block tile, 256 threads = 8 waves (4M x 2N), wave tile 32x64.
// Double-buffered LDS staged with async global->LDS copies.
#define BM 128
#define BN 128
#define BK 32
#define ASTR 40                       // BK + 8 pad (elements)
#define BUFE ((BM + BN) * ASTR)       // elements per LDS buffer

__global__ __launch_bounds__(256)
void gemm_bf16_kernel(const unsigned short* __restrict__ A,    // [M,Kd] bf16
                      const unsigned short* __restrict__ BT,   // [Nc,Kd] bf16
                      const float* __restrict__ bias,          // [Nc]
                      unsigned short* __restrict__ outBf,      // [M,Nc] or null
                      float* __restrict__ outF,                // [M,Nc] or null
                      int M, int Nc, int Kd, int doLeaky) {
    extern __shared__ unsigned short lds[];

    const int tid = threadIdx.x;
    const int lane = tid & 31;
    const int wave = tid >> 5;
    const int waveM = wave >> 1;
    const int waveN = wave & 1;
    const int lhalf = lane >> 4;
    const int lmod  = lane & 15;

    const int ntn = Nc / BN;
    const int m0 = (blockIdx.x / ntn) * BM;
    const int n0 = (blockIdx.x % ntn) * BN;

    // staging geometry: thread covers rows r0 and r0+64, 16B chunk cq
    const int r0 = tid >> 2;
    const int cq = tid & 3;
    const unsigned short* Ab = A + (size_t)(m0 + r0) * Kd + cq * 8;
    const unsigned short* Bb = BT + (size_t)(n0 + r0) * Kd + cq * 8;

    auto stage = [&](int k0, int buf) {
        unsigned short* la = lds + buf * BUFE;
        unsigned short* lb = la + BM * ASTR;
        async_cp16(lds, la + r0 * ASTR + cq * 8, Ab + k0);
        async_cp16(lds, la + (r0 + 64) * ASTR + cq * 8, Ab + (size_t)64 * Kd + k0);
        async_cp16(lds, lb + r0 * ASTR + cq * 8, Bb + k0);
        async_cp16(lds, lb + (r0 + 64) * ASTR + cq * 8, Bb + (size_t)64 * Kd + k0);
    };

    v8f acc[2][4];
    #pragma unroll
    for (int mf = 0; mf < 2; ++mf)
        #pragma unroll
        for (int nf = 0; nf < 4; ++nf)
            #pragma unroll
            for (int v = 0; v < 8; ++v) acc[mf][nf][v] = 0.f;

    stage(0, 0);
    for (int k0 = 0; k0 < Kd; k0 += BK) {
        const int buf = (k0 / BK) & 1;
        wait_async0();
        __syncthreads();
        if (k0 + BK < Kd) stage(k0 + BK, buf ^ 1);

        unsigned short* la = lds + buf * BUFE;
        unsigned short* lb = la + BM * ASTR;
        FragB a[2], b[4];
        #pragma unroll
        for (int mf = 0; mf < 2; ++mf) {
            int r = waveM * 32 + mf * 16 + lmod;
            a[mf].q[0] = *(const U4*)(la + r * ASTR + lhalf * 8);
            a[mf].q[1] = *(const U4*)(la + r * ASTR + 16 + lhalf * 8);
        }
        #pragma unroll
        for (int nf = 0; nf < 4; ++nf) {
            int r = waveN * 64 + nf * 16 + lmod;
            const U4* p = (const U4*)(lb + r * ASTR + lhalf * 8);
            b[nf].q[0] = p[0];
            b[nf].q[1] = p[2];
        }
        #pragma unroll
        for (int mf = 0; mf < 2; ++mf)
            #pragma unroll
            for (int nf = 0; nf < 4; ++nf)
                acc[mf][nf] = __builtin_amdgcn_wmma_f32_16x16x32_bf16(
                    false, a[mf].v, false, b[nf].v, (short)0, acc[mf][nf], false, false);
    }

    #pragma unroll
    for (int mf = 0; mf < 2; ++mf)
        #pragma unroll
        for (int nf = 0; nf < 4; ++nf) {
            int col = n0 + waveN * 64 + nf * 16 + lmod;
            float bv = bias[col];
            #pragma unroll
            for (int v = 0; v < 8; ++v) {
                int row = m0 + waveM * 32 + mf * 16 + v + 8 * lhalf;
                float val = acc[mf][nf][v] + bv;
                if (doLeaky) val = (val >= 0.f) ? val : 0.01f * val;
                if (outF)  outF[(size_t)row * Nc + col] = val;
                if (outBf) outBf[(size_t)row * Nc + col] = f2bf(val);
            }
        }
}

// ---------------- VQ argmin: scores = ||c||^2 - 2 z.c, running argmin ----------------
// 128 z-rows per block (8 waves x 16 rows). Codes streamed in 128-wide chunks.
// z tile staged by async lane copies; codebook tile staged by ONE Tensor-DMA op
// (wave 0), with D# LDS padding reproducing the 40-element padded stride.
__global__ __launch_bounds__(256)
void vq_argmin_kernel(const unsigned short* __restrict__ Zbf,   // [N,E] bf16
                      const unsigned short* __restrict__ CB,    // [K,E] bf16
                      const float* __restrict__ cbn,            // [K]
                      int* __restrict__ outIdx) {               // [N]
    extern __shared__ unsigned short lds[];

    const int tid = threadIdx.x;
    const int lane = tid & 31;
    const int wave = tid >> 5;
    const int lhalf = lane >> 4;
    const int lmod  = lane & 15;
    const int m0 = blockIdx.x * BM;
    const int mrow = wave * 16;

    const int r0 = tid >> 2;
    const int cq = tid & 3;

    const int KSTEPS = EDIM / BK;                 // 16
    const int TOT = (KCODES / 128) * KSTEPS;      // 1024 pipeline steps

    auto stage = [&](int s, int buf) {
        int k0 = (s & (KSTEPS - 1)) * BK;
        int c0 = (s / KSTEPS) * 128;
        unsigned short* la = lds + buf * BUFE;
        async_cp16(lds, la + r0 * ASTR + cq * 8,
                   Zbf + (size_t)(m0 + r0) * EDIM + k0 + cq * 8);
        async_cp16(lds, la + (r0 + 64) * ASTR + cq * 8,
                   Zbf + (size_t)(m0 + r0 + 64) * EDIM + k0 + cq * 8);
#ifdef HAVE_TDM
        if (wave == 0)
            tdm_load_2d((unsigned int)((buf * BUFE + BM * ASTR) * 2),
                        CB + (size_t)c0 * EDIM + k0,
                        /*tileW=*/BK, /*tileH=*/128,
                        /*tensorW=*/EDIM, /*tensorH=*/KCODES,
                        /*stride=*/EDIM,
                        /*pad_interval: 16 DWORDs=*/3, /*pad_amount: 4 DWORDs=*/3);
#else
        unsigned short* lb = la + BM * ASTR;
        async_cp16(lds, lb + r0 * ASTR + cq * 8,
                   CB + (size_t)(c0 + r0) * EDIM + k0 + cq * 8);
        async_cp16(lds, lb + (r0 + 64) * ASTR + cq * 8,
                   CB + (size_t)(c0 + r0 + 64) * EDIM + k0 + cq * 8);
#endif
    };

    float best[8]; int bidx[8];
    #pragma unroll
    for (int v = 0; v < 8; ++v) { best[v] = 3.4e38f; bidx[v] = 0; }

    v8f acc[8];

    stage(0, 0);
    for (int s = 0; s < TOT; ++s) {
        const int buf = s & 1;
        wait_async0();
#ifdef HAVE_TDM
        if (wave == 0) wait_tensor0();
#endif
        __syncthreads();
        if (s + 1 < TOT) stage(s + 1, buf ^ 1);

        if ((s & (KSTEPS - 1)) == 0) {
            #pragma unroll
            for (int nf = 0; nf < 8; ++nf)
                #pragma unroll
                for (int v = 0; v < 8; ++v) acc[nf][v] = 0.f;
        }

        unsigned short* la = lds + buf * BUFE;
        unsigned short* lb = la + BM * ASTR;
        FragB a;
        int r = mrow + lmod;
        a.q[0] = *(const U4*)(la + r * ASTR + lhalf * 8);
        a.q[1] = *(const U4*)(la + r * ASTR + 16 + lhalf * 8);
        #pragma unroll
        for (int nf = 0; nf < 8; ++nf) {
            FragB b;
            const U4* p = (const U4*)(lb + (nf * 16 + lmod) * ASTR + lhalf * 8);
            b.q[0] = p[0];
            b.q[1] = p[2];
            acc[nf] = __builtin_amdgcn_wmma_f32_16x16x32_bf16(
                false, a.v, false, b.v, (short)0, acc[nf], false, false);
        }

        if ((s & (KSTEPS - 1)) == KSTEPS - 1) {
            int c0 = (s / KSTEPS) * 128;
            #pragma unroll
            for (int nf = 0; nf < 8; ++nf) {
                int code = c0 + nf * 16 + lmod;
                float cn = cbn[code];
                #pragma unroll
                for (int v = 0; v < 8; ++v) {
                    float score = cn - 2.0f * acc[nf][v];
                    if (score < best[v]) { best[v] = score; bidx[v] = code; }
                }
            }
        }
    }

    #pragma unroll
    for (int v = 0; v < 8; ++v) {
        float bv = best[v]; int bi = bidx[v];
        #pragma unroll
        for (int m = 8; m >= 1; m >>= 1) {
            float ov = __shfl_xor(bv, m, 32);
            int   oi = __shfl_xor(bi, m, 32);
            if (ov < bv) { bv = ov; bi = oi; }
        }
        if (lmod == 0) outIdx[m0 + mrow + v + 8 * lhalf] = bi;
    }
}

// ---------------- launcher ----------------
extern "C" void kernel_launch(void* const* d_in, const int* in_sizes, int n_in,
                              void* d_out, int out_size, void* d_ws, size_t ws_size,
                              hipStream_t stream) {
    const float* x   = (const float*)d_in[0];
    const float* ew1 = (const float*)d_in[1];
    const float* eb1 = (const float*)d_in[2];
    const float* ew2 = (const float*)d_in[3];
    const float* eb2 = (const float*)d_in[4];
    const float* cb  = (const float*)d_in[5];
    const float* dw1 = (const float*)d_in[6];
    const float* db1 = (const float*)d_in[7];
    const float* dw2 = (const float*)d_in[8];
    const float* db2 = (const float*)d_in[9];
    float* out = (float*)d_out;

    char* ws = (char*)d_ws;
    size_t o = 0;
    auto alloc = [&](size_t bytes) { size_t r = o; o += (bytes + 255) & ~(size_t)255; return r; };

    unsigned short* xbf   = (unsigned short*)(ws + alloc((size_t)NROWS * INDIM * 2));
    unsigned short* ew1T  = (unsigned short*)(ws + alloc((size_t)INDIM * INDIM * 2));
    unsigned short* ew2T  = (unsigned short*)(ws + alloc((size_t)EDIM * INDIM * 2));
    unsigned short* dw1T  = (unsigned short*)(ws + alloc((size_t)INDIM * EDIM * 2));
    unsigned short* dw2T  = (unsigned short*)(ws + alloc((size_t)INDIM * INDIM * 2));
    unsigned short* cbbf  = (unsigned short*)(ws + alloc((size_t)KCODES * EDIM * 2));
    float*          cbn   = (float*)(ws + alloc((size_t)KCODES * 4));
    unsigned short* hbf   = (unsigned short*)(ws + alloc((size_t)NROWS * INDIM * 2));
    float*          zf    = (float*)(ws + alloc((size_t)NROWS * EDIM * 4));
    unsigned short* zbf   = (unsigned short*)(ws + alloc((size_t)NROWS * EDIM * 2));
    unsigned short* zqbf  = (unsigned short*)(ws + alloc((size_t)NROWS * EDIM * 2));
    int*            idx   = (int*)(ws + alloc((size_t)NROWS * 4));
    float*          loss  = (float*)(ws + alloc(256));

    const size_t shmem = (size_t)2 * BUFE * 2;  // two 20.0KB buffers = 40960 B

    zero_kernel<<<1, 1, 0, stream>>>(loss);
    f2bf_kernel<<<((size_t)NROWS * INDIM) / (256 * 8), 256, 0, stream>>>(x, xbf, (long)NROWS * INDIM);
    f2bf_kernel<<<((size_t)KCODES * EDIM) / (256 * 8), 256, 0, stream>>>(cb, cbbf, (long)KCODES * EDIM);
    transpose_f2bf_kernel<<<(INDIM / 32) * (INDIM / 32), 256, 0, stream>>>(ew1, ew1T, INDIM, INDIM);
    transpose_f2bf_kernel<<<(INDIM / 32) * (EDIM / 32), 256, 0, stream>>>(ew2, ew2T, INDIM, EDIM);
    transpose_f2bf_kernel<<<(EDIM / 32) * (INDIM / 32), 256, 0, stream>>>(dw1, dw1T, EDIM, INDIM);
    transpose_f2bf_kernel<<<(INDIM / 32) * (INDIM / 32), 256, 0, stream>>>(dw2, dw2T, INDIM, INDIM);
    cbnorm_kernel<<<KCODES / 8, 256, 0, stream>>>(cb, cbn);

    gemm_bf16_kernel<<<(NROWS / BM) * (INDIM / BN), 256, shmem, stream>>>(
        xbf, ew1T, eb1, hbf, nullptr, NROWS, INDIM, INDIM, 1);
    gemm_bf16_kernel<<<(NROWS / BM) * (EDIM / BN), 256, shmem, stream>>>(
        hbf, ew2T, eb2, zbf, zf, NROWS, EDIM, INDIM, 0);

    vq_argmin_kernel<<<NROWS / BM, 256, shmem, stream>>>(zbf, cbbf, cbn, idx);
    gather_loss_kernel<<<NROWS / 8, 256, 0, stream>>>(idx, cb, cbbf, zf, zqbf, loss);

    gemm_bf16_kernel<<<(NROWS / BM) * (INDIM / BN), 256, shmem, stream>>>(
        zqbf, dw1T, db1, hbf, nullptr, NROWS, INDIM, EDIM, 1);
    gemm_bf16_kernel<<<(NROWS / BM) * (INDIM / BN), 256, shmem, stream>>>(
        hbf, dw2T, db2, nullptr, out, NROWS, INDIM, INDIM, 0);

    finalize_kernel<<<1, 1, 0, stream>>>(loss, out + (size_t)NROWS * INDIM);
    (void)in_sizes; (void)n_in; (void)out_size; (void)ws_size;
}
